// TransformerBlock_75015898792266
// MI455X (gfx1250) — matrix operand order repaired and checked by reading, not compiled, not run on hardware
//
#include <hip/hip_runtime.h>

typedef unsigned short u16;
typedef __attribute__((ext_vector_type(16))) __bf16 v16bf;
typedef __attribute__((ext_vector_type(8)))  float  v8f;
typedef __attribute__((__vector_size__(4 * sizeof(int)))) int v4i_t;

union Frag16 { uint4 q[2]; v16bf v; };

// ---- gfx1250 async global->LDS path (guarded: falls back if not declared) ---
#ifndef USE_ASYNC_LDS
#if defined(__has_builtin)
#if __has_builtin(__builtin_amdgcn_global_load_async_to_lds_b128) && \
    __has_builtin(__builtin_amdgcn_s_wait_asynccnt)
#define USE_ASYNC_LDS 1
#else
#define USE_ASYNC_LDS 0
#endif
#else
#define USE_ASYNC_LDS 0
#endif
#endif

#if USE_ASYNC_LDS
__device__ __forceinline__ void async_copy16(const u16* g, u16* l) {
  __builtin_amdgcn_global_load_async_to_lds_b128(
      (__attribute__((address_space(1))) v4i_t*)g,
      (__attribute__((address_space(3))) v4i_t*)l, 0, 0);
}
#endif

__device__ __forceinline__ u16 f2bf(float f) {
  unsigned int u = __float_as_uint(f);
  u += 0x7FFFu + ((u >> 16) & 1u);           // round-to-nearest-even
  return (u16)(u >> 16);
}

__device__ __forceinline__ v8f wmma_bf16(Frag16 a, Frag16 b, v8f c) {
  return __builtin_amdgcn_wmma_f32_16x16x32_bf16(
      /*neg_a=*/false, a.v, /*neg_b=*/false, b.v,
      /*c_mod=*/(short)0, c, /*reuse_a=*/false, /*reuse_b=*/false);
}

// ---------------------------------------------------------------------------
// Weight convert + transpose: W fp32 [K][N] -> Wt bf16 [N][K]
// ---------------------------------------------------------------------------
__global__ __launch_bounds__(256) void wt_kernel(const float* __restrict__ W,
                                                 u16* __restrict__ Wt,
                                                 int K, int N) {
  size_t idx = (size_t)blockIdx.x * 256 + threadIdx.x;
  int k = (int)(idx / N);
  int n = (int)(idx % N);
  Wt[(size_t)n * K + k] = f2bf(W[idx]);
}

// ---------------------------------------------------------------------------
// LayerNorm: fp32 [rows][D] -> bf16 [rows][D]
// ---------------------------------------------------------------------------
__global__ __launch_bounds__(256) void ln_bf16(const float* __restrict__ X,
                                               const float* __restrict__ g,
                                               const float* __restrict__ b,
                                               u16* __restrict__ H, int D) {
  __shared__ float red[256];
  int row = blockIdx.x, tid = threadIdx.x;
  const float* xr = X + (size_t)row * D;

  float s = 0.f;
  for (int i = tid; i < D; i += 256) s += xr[i];
  red[tid] = s; __syncthreads();
  for (int off = 128; off > 0; off >>= 1) {
    if (tid < off) red[tid] += red[tid + off];
    __syncthreads();
  }
  float mu = red[0] / D;
  __syncthreads();

  float s2 = 0.f;
  for (int i = tid; i < D; i += 256) { float d = xr[i] - mu; s2 += d * d; }
  red[tid] = s2; __syncthreads();
  for (int off = 128; off > 0; off >>= 1) {
    if (tid < off) red[tid] += red[tid + off];
    __syncthreads();
  }
  float rstd = rsqrtf(red[0] / D + 1e-5f);

  for (int i = tid; i < D; i += 256)
    H[(size_t)row * D + i] = f2bf((xr[i] - mu) * rstd * g[i] + b[i]);
}

// ---------------------------------------------------------------------------
// WMMA GEMM, 128x128 block tile, double-buffered LDS, K-step 32.
// Block = 256 thr (8 waves, 2x4): wave owns 64x32 of output = 8 WMMA / K-step.
// ---------------------------------------------------------------------------
__device__ __forceinline__ void gemm_compute(const u16* Asb, const u16* Bsb,
                                             v8f acc[4][2], int wm, int wn,
                                             int ll, int hl) {
  Frag16 fa[4], fb[2];
#pragma unroll
  for (int im = 0; im < 4; ++im) {
    const u16* ap = &Asb[(64 * wm + 16 * im + ll) * 32 + hl * 8];
    fa[im].q[0] = *(const uint4*)ap;
    fa[im].q[1] = *(const uint4*)(ap + 16);
  }
#pragma unroll
  for (int in = 0; in < 2; ++in) {
    const u16* bp = &Bsb[(32 * wn + 16 * in + ll) * 32 + hl * 8];
    fb[in].q[0] = *(const uint4*)bp;
    fb[in].q[1] = *(const uint4*)(bp + 16);
  }
#pragma unroll
  for (int im = 0; im < 4; ++im)
#pragma unroll
    for (int in = 0; in < 2; ++in)
      acc[im][in] = wmma_bf16(fa[im], fb[in], acc[im][in]);
}

__global__ __launch_bounds__(256) void gemm_bf16(
    const u16* __restrict__ A, const u16* __restrict__ Bt,
    const float* __restrict__ bias, const float* __restrict__ residual,
    float* __restrict__ Cf, u16* __restrict__ Cb,
    int M, int N, int K, int relu) {
  __shared__ __align__(16) u16 As[2][128 * 32];
  __shared__ __align__(16) u16 Bs[2][128 * 32];

  int m0 = blockIdx.y * 128, n0 = blockIdx.x * 128;
  int t = threadIdx.x, lane = t & 31, w = t >> 5;
  int wm = w >> 2, wn = w & 3;              // 2 x 4 wave grid
  int ll = lane & 15, hl = lane >> 4;

  // staging: 512 chunks (16B) per matrix per tile -> 2 chunks/thread
  int c0 = t, c1 = t + 256;
  int r0 = c0 >> 2, s0c = (c0 & 3) * 8;
  int r1 = c1 >> 2, s1c = (c1 & 3) * 8;

  const u16* Ag = A + (size_t)m0 * K;
  const u16* Bg = Bt + (size_t)n0 * K;

  v8f acc[4][2];
#pragma unroll
  for (int im = 0; im < 4; ++im)
#pragma unroll
    for (int in = 0; in < 2; ++in) acc[im][in] = (v8f){};

#if USE_ASYNC_LDS
  // direct-to-LDS async double buffering (no VGPR round trip)
  {
    async_copy16(&Ag[(size_t)r0 * K + s0c], &As[0][r0 * 32 + s0c]);
    async_copy16(&Ag[(size_t)r1 * K + s1c], &As[0][r1 * 32 + s1c]);
    async_copy16(&Bg[(size_t)r0 * K + s0c], &Bs[0][r0 * 32 + s0c]);
    async_copy16(&Bg[(size_t)r1 * K + s1c], &Bs[0][r1 * 32 + s1c]);
    __builtin_amdgcn_s_wait_asynccnt(0);
    __syncthreads();
  }
  int cur = 0;
  for (int k0 = 32; k0 < K; k0 += 32) {
    int nb = cur ^ 1;
    async_copy16(&Ag[(size_t)r0 * K + k0 + s0c], &As[nb][r0 * 32 + s0c]);
    async_copy16(&Ag[(size_t)r1 * K + k0 + s1c], &As[nb][r1 * 32 + s1c]);
    async_copy16(&Bg[(size_t)r0 * K + k0 + s0c], &Bs[nb][r0 * 32 + s0c]);
    async_copy16(&Bg[(size_t)r1 * K + k0 + s1c], &Bs[nb][r1 * 32 + s1c]);
    gemm_compute(As[cur], Bs[cur], acc, wm, wn, ll, hl);
    __builtin_amdgcn_s_wait_asynccnt(0);
    __syncthreads();
    cur = nb;
  }
  gemm_compute(As[cur], Bs[cur], acc, wm, wn, ll, hl);
#else
  // register double buffering: global loads overlap WMMA on current tile
  uint4 ra0, ra1, rb0, rb1;
  ra0 = *(const uint4*)&Ag[(size_t)r0 * K + s0c];
  ra1 = *(const uint4*)&Ag[(size_t)r1 * K + s1c];
  rb0 = *(const uint4*)&Bg[(size_t)r0 * K + s0c];
  rb1 = *(const uint4*)&Bg[(size_t)r1 * K + s1c];
  *(uint4*)&As[0][r0 * 32 + s0c] = ra0;
  *(uint4*)&As[0][r1 * 32 + s1c] = ra1;
  *(uint4*)&Bs[0][r0 * 32 + s0c] = rb0;
  *(uint4*)&Bs[0][r1 * 32 + s1c] = rb1;
  __syncthreads();
  int cur = 0;
  for (int k0 = 32; k0 < K; k0 += 32) {
    ra0 = *(const uint4*)&Ag[(size_t)r0 * K + k0 + s0c];
    ra1 = *(const uint4*)&Ag[(size_t)r1 * K + k0 + s1c];
    rb0 = *(const uint4*)&Bg[(size_t)r0 * K + k0 + s0c];
    rb1 = *(const uint4*)&Bg[(size_t)r1 * K + k0 + s1c];
    gemm_compute(As[cur], Bs[cur], acc, wm, wn, ll, hl);
    int nb = cur ^ 1;
    *(uint4*)&As[nb][r0 * 32 + s0c] = ra0;
    *(uint4*)&As[nb][r1 * 32 + s1c] = ra1;
    *(uint4*)&Bs[nb][r0 * 32 + s0c] = rb0;
    *(uint4*)&Bs[nb][r1 * 32 + s1c] = rb1;
    __syncthreads();
    cur = nb;
  }
  gemm_compute(As[cur], Bs[cur], acc, wm, wn, ll, hl);
#endif

  // Epilogue: C/D layout -> element (M = hl*8 + i, N = ll) per lane
#pragma unroll
  for (int im = 0; im < 4; ++im) {
    int rbase = m0 + 64 * wm + 16 * im + hl * 8;
#pragma unroll
    for (int in = 0; in < 2; ++in) {
      int col = n0 + 32 * wn + 16 * in + ll;
      float bval = bias ? bias[col] : 0.f;
#pragma unroll
      for (int i = 0; i < 8; ++i) {
        int r = rbase + i;
        float v = acc[im][in][i] + bval;
        if (relu) v = fmaxf(v, 0.f);
        if (residual) v += residual[(size_t)r * N + col];
        if (Cb) Cb[(size_t)r * N + col] = f2bf(v);
        else    Cf[(size_t)r * N + col] = v;
      }
    }
  }
}

// ---------------------------------------------------------------------------
// Flash attention (causal), dk=64, H=16. Block = 128 thr (4 waves), 64 query
// rows per block (16 per wave), key tiles of 32. Grid = (S/64, B*H).
// ---------------------------------------------------------------------------
__global__ __launch_bounds__(128) void attn_kernel(
    const u16* __restrict__ Qg, const u16* __restrict__ Kg,
    const u16* __restrict__ Vg, u16* __restrict__ Ctx, int S, int Dm) {
  __shared__ __align__(16) u16 Qs[64 * 64];
  __shared__ __align__(16) u16 Ks[32 * 64];
  __shared__ __align__(16) u16 Vt[64 * 32];      // transposed: [dk][key]
  __shared__ __align__(16) u16 Pt[4 * 16 * 32];  // per-wave P buffer

  int q0 = blockIdx.x * 64;
  int bh = blockIdx.y;
  int b = bh >> 4, h = bh & 15;
  size_t tokBase = (size_t)b * S;

  int t = threadIdx.x, lane = t & 31, w = t >> 5;
  int ll = lane & 15, hl = lane >> 4;
  const float NEG_INF = -__builtin_inff();

  // stage Q tile 64x64
#pragma unroll
  for (int i = 0; i < 4; ++i) {
    int cid = t + 128 * i;
    int row = cid >> 3, ch = (cid & 7) * 8;
    *(uint4*)&Qs[row * 64 + ch] =
        *(const uint4*)&Qg[(tokBase + q0 + row) * (size_t)Dm + h * 64 + ch];
  }
  __syncthreads();

  // per-wave Q fragments (two K=32 steps over dk=64), kept in registers
  Frag16 fq[2];
#pragma unroll
  for (int ks = 0; ks < 2; ++ks) {
    const u16* p = &Qs[(w * 16 + ll) * 64 + ks * 32 + hl * 8];
    fq[ks].q[0] = *(const uint4*)p;
    fq[ks].q[1] = *(const uint4*)(p + 16);
  }

  v8f o[4] = {{}, {}, {}, {}};
  float mrow[8], lrow[8];
#pragma unroll
  for (int i = 0; i < 8; ++i) { mrow[i] = NEG_INF; lrow[i] = 0.f; }

  int qrow_w = q0 + w * 16;

  for (int j = 0; j <= q0 + 32; j += 32) {
    __syncthreads();   // previous tile fully consumed before restaging
    // stage K tile 32x64 (row-major: key x dk)
#pragma unroll
    for (int i = 0; i < 2; ++i) {
      int cid = t + 128 * i;
      int row = cid >> 3, ch = (cid & 7) * 8;
      *(uint4*)&Ks[row * 64 + ch] =
          *(const uint4*)&Kg[(tokBase + j + row) * (size_t)Dm + h * 64 + ch];
    }
    // stage V transposed: Vt[d][key]
    {
      int d = t >> 1, kb = (t & 1) * 16;
#pragma unroll
      for (int kk = 0; kk < 16; ++kk) {
        int key = kb + kk;
        Vt[d * 32 + key] = Vg[(tokBase + j + key) * (size_t)Dm + h * 64 + d];
      }
    }
    // prefetch next K/V key tile into cache (global_prefetch_b8)
    {
      int jn = j + 32;
      if (jn <= q0 + 32) {
        int rr = t & 31, cc = (t >> 5) * 16;
        __builtin_prefetch(&Kg[(tokBase + jn + rr) * (size_t)Dm + h * 64 + cc], 0, 0);
        __builtin_prefetch(&Vg[(tokBase + jn + rr) * (size_t)Dm + h * 64 + cc], 0, 0);
      }
    }
    __syncthreads();

    // scores S = Q K^T : two key n-tiles x two dk k-steps
    v8f s0 = {}, s1 = {};
#pragma unroll
    for (int ks = 0; ks < 2; ++ks) {
      Frag16 fk0, fk1;
      const u16* kp0 = &Ks[ll * 64 + ks * 32 + hl * 8];
      fk0.q[0] = *(const uint4*)kp0;
      fk0.q[1] = *(const uint4*)(kp0 + 16);
      const u16* kp1 = &Ks[(16 + ll) * 64 + ks * 32 + hl * 8];
      fk1.q[0] = *(const uint4*)kp1;
      fk1.q[1] = *(const uint4*)(kp1 + 16);
      s0 = wmma_bf16(fq[ks], fk0, s0);
      s1 = wmma_bf16(fq[ks], fk1, s1);
    }

    // online softmax (row = hl*8+i; this lane's keys: j+ll, j+16+ll)
    u16* pw = &Pt[w * 16 * 32];
    float aex[8];
#pragma unroll
    for (int i = 0; i < 8; ++i) {
      int qg = qrow_w + hl * 8 + i;
      float a = s0[i] * 0.125f;      // 1/sqrt(64)
      float c = s1[i] * 0.125f;
      if (j + ll > qg)      a = NEG_INF;
      if (j + 16 + ll > qg) c = NEG_INF;
      float mx = fmaxf(a, c);
#pragma unroll
      for (int mm = 1; mm < 16; mm <<= 1)
        mx = fmaxf(mx, __shfl_xor(mx, mm, 32));
      float mn = fmaxf(mrow[i], mx);
      float al = __expf(mrow[i] - mn);
      float p0 = __expf(a - mn);
      float p1 = __expf(c - mn);
      float ps = p0 + p1;
#pragma unroll
      for (int mm = 1; mm < 16; mm <<= 1)
        ps += __shfl_xor(ps, mm, 32);
      lrow[i] = lrow[i] * al + ps;
      mrow[i] = mn;
      aex[i] = al;
      int r = hl * 8 + i;
      pw[r * 32 + ll] = f2bf(p0);
      pw[r * 32 + 16 + ll] = f2bf(p1);
    }

    // rescale accumulators
#pragma unroll
    for (int tn = 0; tn < 4; ++tn)
#pragma unroll
      for (int i = 0; i < 8; ++i) o[tn][i] *= aex[i];

    // P fragment (A-layout) from per-wave LDS buffer
    Frag16 fp;
    const u16* pp = &pw[ll * 32 + hl * 8];
    fp.q[0] = *(const uint4*)pp;
    fp.q[1] = *(const uint4*)(pp + 16);

    // O += P x V  (four dk n-tiles)
#pragma unroll
    for (int tn = 0; tn < 4; ++tn) {
      Frag16 fv;
      const u16* vp = &Vt[(tn * 16 + ll) * 32 + hl * 8];
      fv.q[0] = *(const uint4*)vp;
      fv.q[1] = *(const uint4*)(vp + 16);
      o[tn] = wmma_bf16(fp, fv, o[tn]);
    }
  }

  // normalize + store context (bf16, token-major [t][h*64+d])
#pragma unroll
  for (int i = 0; i < 8; ++i) {
    int r = qrow_w + hl * 8 + i;
    float inv = 1.0f / lrow[i];
    size_t base = (tokBase + r) * (size_t)Dm + h * 64;
#pragma unroll
    for (int tn = 0; tn < 4; ++tn)
      Ctx[base + tn * 16 + ll] = f2bf(o[tn][i] * inv);
  }
}

// ---------------------------------------------------------------------------
// Host-side orchestration
// ---------------------------------------------------------------------------
extern "C" void kernel_launch(void* const* d_in, const int* in_sizes, int n_in,
                              void* d_out, int out_size, void* d_ws, size_t ws_size,
                              hipStream_t stream) {
  const int D = 1024, FF = 4096, S = 2048, B = 2;
  const int TOK = B * S;   // 4096

  const float* x   = (const float*)d_in[0];
  // d_in[1] = mask (causality computed analytically)
  const float* Wq  = (const float*)d_in[2];
  const float* bq  = (const float*)d_in[3];
  const float* Wk  = (const float*)d_in[4];
  const float* bk  = (const float*)d_in[5];
  const float* Wv  = (const float*)d_in[6];
  const float* bv  = (const float*)d_in[7];
  const float* Wo  = (const float*)d_in[8];
  const float* bo  = (const float*)d_in[9];
  const float* W1  = (const float*)d_in[10];
  const float* b1  = (const float*)d_in[11];
  const float* W2  = (const float*)d_in[12];
  const float* b2  = (const float*)d_in[13];
  const float* g1  = (const float*)d_in[14];
  const float* be1 = (const float*)d_in[15];
  const float* g2  = (const float*)d_in[16];
  const float* be2 = (const float*)d_in[17];

  char* ws = (char*)d_ws;
  const size_t MB = 1u << 20;
  u16*   WqT = (u16*)(ws + 0 * MB);    // 2 MB each
  u16*   WkT = (u16*)(ws + 2 * MB);
  u16*   WvT = (u16*)(ws + 4 * MB);
  u16*   WoT = (u16*)(ws + 6 * MB);
  u16*   W1T = (u16*)(ws + 8 * MB);    // 8 MB
  u16*   W2T = (u16*)(ws + 16 * MB);   // 8 MB
  u16*   Hb  = (u16*)(ws + 24 * MB);   // 8 MB (LN1 out, reused as LN2 out)
  u16*   Qb  = (u16*)(ws + 32 * MB);   // 8 MB
  u16*   Kb  = (u16*)(ws + 40 * MB);   // 8 MB
  u16*   Vb  = (u16*)(ws + 48 * MB);   // 8 MB
  u16*   Cx  = (u16*)(ws + 56 * MB);   // 8 MB (attention context)
  float* X1  = (float*)(ws + 64 * MB); // 16 MB (post-attn residual)
  u16*   A1  = (u16*)(ws + 80 * MB);   // 32 MB (FFN hidden)

  // 1) weight convert + transpose to bf16 [N][K]
  wt_kernel<<<(D * D) / 256, 256, 0, stream>>>(Wq, WqT, D, D);
  wt_kernel<<<(D * D) / 256, 256, 0, stream>>>(Wk, WkT, D, D);
  wt_kernel<<<(D * D) / 256, 256, 0, stream>>>(Wv, WvT, D, D);
  wt_kernel<<<(D * D) / 256, 256, 0, stream>>>(Wo, WoT, D, D);
  wt_kernel<<<(D * FF) / 256, 256, 0, stream>>>(W1, W1T, D, FF);
  wt_kernel<<<(FF * D) / 256, 256, 0, stream>>>(W2, W2T, FF, D);

  // 2) LN1 -> bf16
  ln_bf16<<<TOK, 256, 0, stream>>>(x, g1, be1, Hb, D);

  // 3) QKV projections (WMMA GEMM, bf16 out)
  dim3 gD(D / 128, TOK / 128);     // (8, 32)
  gemm_bf16<<<gD, 256, 0, stream>>>(Hb, WqT, bq, nullptr, nullptr, Qb, TOK, D, D, 0);
  gemm_bf16<<<gD, 256, 0, stream>>>(Hb, WkT, bk, nullptr, nullptr, Kb, TOK, D, D, 0);
  gemm_bf16<<<gD, 256, 0, stream>>>(Hb, WvT, bv, nullptr, nullptr, Vb, TOK, D, D, 0);

  // 4) causal flash attention
  attn_kernel<<<dim3(S / 64, B * 16), 128, 0, stream>>>(Qb, Kb, Vb, Cx, S, D);

  // 5) O-proj + residual -> X1 (fp32)
  gemm_bf16<<<gD, 256, 0, stream>>>(Cx, WoT, bo, x, X1, nullptr, TOK, D, D, 0);

  // 6) LN2 -> bf16
  ln_bf16<<<TOK, 256, 0, stream>>>(X1, g2, be2, Hb, D);

  // 7) FFN1 + ReLU (bf16 out)
  gemm_bf16<<<dim3(FF / 128, TOK / 128), 256, 0, stream>>>(
      Hb, W1T, b1, nullptr, nullptr, A1, TOK, FF, D, 1);

  // 8) FFN2 + residual -> d_out (fp32)
  gemm_bf16<<<gD, 256, 0, stream>>>(A1, W2T, b2, X1, (float*)d_out, nullptr,
                                    TOK, D, FF, 0);
}
